// ExpFilter_61143154426043
// MI455X (gfx1250) — matrix operand order repaired and checked by reading, not compiled, not run on hardware
//
#include <hip/hip_runtime.h>

typedef _Float16 v16h __attribute__((ext_vector_type(16)));
typedef _Float16 v8h  __attribute__((ext_vector_type(8)));
typedef float    v8f  __attribute__((ext_vector_type(8)));

#define T_DIM   4096
#define B_DIM   32
#define I_DIM   512
#define O_DIM   512
#define P_EMA   0.1f
#define HALO    16              // 0.1^17 ~ 1e-17: below fp32 resolution
#define T_OUT   48              // output t-rows per block
#define ROWS    (T_OUT + HALO)  // 64 y-rows (4 M-tiles)
#define O_BLK   64              // o-columns per block (4 N-tiles)
#define KC      64              // K-chunk staged in LDS
#define KP      (KC + 8)        // padded LDS row stride (f16 elems) -> conflict-free
#define YS      (O_BLK + 1)     // y overlay stride (f32 elems)

// f32 -> (hi, lo) f16 split for 8 values
__device__ __forceinline__ void split8(v8h& h, v8h& l, float4 a, float4 b) {
    float v[8] = {a.x, a.y, a.z, a.w, b.x, b.y, b.z, b.w};
#pragma unroll
    for (int i = 0; i < 8; ++i) {
        _Float16 hh = (_Float16)v[i];
        h[i] = hh;
        l[i] = (_Float16)(v[i] - (float)hh);
    }
}

__device__ __forceinline__ v16h cat8(v8h a, v8h b) {
    return __builtin_shufflevector(a, b, 0, 1, 2, 3, 4, 5, 6, 7,
                                         8, 9, 10, 11, 12, 13, 14, 15);
}

__global__ __launch_bounds__(256)
void fused_linear_ema_kernel(const float* __restrict__ x,
                             const float* __restrict__ w,
                             const float* __restrict__ bias,
                             float* __restrict__ out)
{
    // LDS plan: Ahi | Alo | Bhi | Blo  (each ROWS*KP f16 = 9216 B) = 36864 B total.
    // The 64x65 f32 y-tile (16640 B) overlays the dead Ahi/Alo region after compute.
    __shared__ __align__(16) char smem[4 * ROWS * KP * sizeof(_Float16)];
    _Float16* Ahi = (_Float16*)smem;
    _Float16* Alo = Ahi + ROWS * KP;
    _Float16* Bhi = Alo + ROWS * KP;
    _Float16* Blo = Bhi + O_BLK * KP;
    float*    ylds = (float*)smem;      // overlay

    const int tid  = threadIdx.x;
    const int lane = tid & 31;
    const int wave = tid >> 5;      // 0..7
    const int mt   = wave & 3;      // M-tile 0..3
    const int ngrp = wave >> 2;     // N-group 0..1 (each = 2 N-tiles)
    const int l15  = lane & 15;
    const int hi16 = lane >> 4;

    const int o_base = blockIdx.x * O_BLK;
    const int t_base = blockIdx.y * T_OUT;
    const int b      = blockIdx.z;

    // ---- staging addresses (one row of 16 floats per thread for A and for W) ----
    const int srow = tid >> 2;               // 0..63
    const int kseg = (tid & 3) << 4;         // 0,16,32,48
    int tg = t_base - HALO + srow;
    int tc = tg < 0 ? 0 : (tg >= T_DIM ? T_DIM - 1 : tg);   // clamped halo row
    const float* __restrict__ aptr = x + ((size_t)tc * B_DIM + b) * I_DIM + kseg;
    const float* __restrict__ wptr = w + (size_t)(o_base + srow) * I_DIM + kseg;
    const int stA = srow * KP + kseg;

    // ---- fragment LDS offsets for this wave/lane ----
    const int m_row = mt * 16 + l15;                 // A row in LDS
    const int aoff0 = m_row * KP + (hi16 << 3);      // + k0 at use site

    v8f acc0 = {}, acc1 = {};

    for (int kk = 0; kk < I_DIM; kk += KC) {
        // issue global loads early (independent of LDS)
        float4 a0 = *(const float4*)(aptr + kk +  0);
        float4 a1 = *(const float4*)(aptr + kk +  4);
        float4 a2 = *(const float4*)(aptr + kk +  8);
        float4 a3 = *(const float4*)(aptr + kk + 12);
        float4 w0 = *(const float4*)(wptr + kk +  0);
        float4 w1 = *(const float4*)(wptr + kk +  4);
        float4 w2 = *(const float4*)(wptr + kk +  8);
        float4 w3 = *(const float4*)(wptr + kk + 12);

        __syncthreads();   // previous chunk's fragment reads complete

        v8h h, l;
        split8(h, l, a0, a1); *(v8h*)&Ahi[stA]     = h; *(v8h*)&Alo[stA]     = l;
        split8(h, l, a2, a3); *(v8h*)&Ahi[stA + 8] = h; *(v8h*)&Alo[stA + 8] = l;
        split8(h, l, w0, w1); *(v8h*)&Bhi[stA]     = h; *(v8h*)&Blo[stA]     = l;
        split8(h, l, w2, w3); *(v8h*)&Bhi[stA + 8] = h; *(v8h*)&Blo[stA + 8] = l;

        __syncthreads();   // staged chunk visible

#pragma unroll
        for (int ks = 0; ks < 2; ++ks) {
            const int k0 = ks << 5;          // 0, 32
            // A frag: lane<16 -> K = k0+[0..7 | 16..23]; lane>=16 -> +8
            const int ao = aoff0 + k0;
            v16h ah = cat8(*(const v8h*)&Ahi[ao], *(const v8h*)&Ahi[ao + 16]);
            v16h al = cat8(*(const v8h*)&Alo[ao], *(const v8h*)&Alo[ao + 16]);
#pragma unroll
            for (int nt2 = 0; nt2 < 2; ++nt2) {
                // B frag: lane holds col o, K = k0+[0..7 | 16..23]; lane>=16 -> +8
                const int o_row = (ngrp << 5) + (nt2 << 4) + l15;
                const int bo = o_row * KP + k0 + (hi16 << 3);
                v16h bh = cat8(*(const v8h*)&Bhi[bo], *(const v8h*)&Bhi[bo + 16]);
                v16h bl = cat8(*(const v8h*)&Blo[bo], *(const v8h*)&Blo[bo + 16]);

                v8f& acc = nt2 ? acc1 : acc0;
                acc = __builtin_amdgcn_wmma_f32_16x16x32_f16(false, ah, false, bh,
                                                             (short)0, acc, false, false);
                acc = __builtin_amdgcn_wmma_f32_16x16x32_f16(false, ah, false, bl,
                                                             (short)0, acc, false, false);
                acc = __builtin_amdgcn_wmma_f32_16x16x32_f16(false, al, false, bh,
                                                             (short)0, acc, false, false);
            }
        }
    }

    __syncthreads();   // all fragment reads done; safe to overlay y on Ahi/Alo

    // ---- bias + stage y tile (64 x 64 f32) ----
    // C/D layout: lane<16 -> col=lane, M=vgpr; lane>=16 -> col=lane-16, M=8+vgpr
    const int rbase = mt * 16 + (hi16 << 3);
#pragma unroll
    for (int nt2 = 0; nt2 < 2; ++nt2) {
        const int col  = (ngrp << 5) + (nt2 << 4) + l15;
        const float bv = bias[o_base + col];
        const v8f& acc = nt2 ? acc1 : acc0;
#pragma unroll
        for (int r = 0; r < 8; ++r)
            ylds[(rbase + r) * YS + col] = acc[r] + bv;
    }

    __syncthreads();

    // ---- 17-tap FIR == EMA (exact to fp32): out[t] = sum_k p^k y[t-k] ----
    for (int j = tid; j < T_OUT * O_BLK; j += 256) {
        const int tl = j >> 6;              // 0..47
        const int ol = j & (O_BLK - 1);     // 0..63
        const int t  = t_base + tl;
        if (t < T_DIM) {
            const int kmax = t < HALO ? t : HALO;
            float s = 0.0f, wk = 1.0f;
#pragma unroll
            for (int k = 0; k <= HALO; ++k) {
                if (k <= kmax)
                    s = fmaf(wk, ylds[(tl + HALO - k) * YS + ol], s);
                wk *= P_EMA;
            }
            out[((size_t)t * B_DIM + b) * O_DIM + o_base + ol] = s;
        }
    }
}

extern "C" void kernel_launch(void* const* d_in, const int* in_sizes, int n_in,
                              void* d_out, int out_size, void* d_ws, size_t ws_size,
                              hipStream_t stream) {
    const float* x    = (const float*)d_in[0];
    const float* w    = (const float*)d_in[1];
    const float* bias = (const float*)d_in[2];
    float* out        = (float*)d_out;

    dim3 grid(O_DIM / O_BLK,                    // 8 o-blocks
              (T_DIM + T_OUT - 1) / T_OUT,      // 86 t-blocks
              B_DIM);                           // 32 batches
    fused_linear_ema_kernel<<<grid, 256, 0, stream>>>(x, w, bias, out);
}
